// AdaptiveMemorySystem_68066641707193
// MI455X (gfx1250) — matrix operand order repaired and checked by reading, not compiled, not run on hardware
//
#include <hip/hip_runtime.h>

typedef __attribute__((ext_vector_type(16))) _Float16 v16h;
typedef __attribute__((ext_vector_type(8)))  float    v8f;

#define THREADS 256
#define BM 128
#define BN 128
#define BK 32
#define LDH 40   // LDS row stride in halves (80 bytes) -> conflict-free b128 reads

union Stage16 { _Float16 h[16]; uint4 q[2]; };

// ---------------------------------------------------------------------------
// Generic tiled GEMM: C[z] = epilogue(alpha * A[z] @ B[z] + bias + add)
//   A: [M,K] f32, lda; TRANSB? B stored [N,K] (row n, k contiguous) : [K,N]
//   SCALE_ACC: C[r,c] += scale[r*sstride+scol] * v   (else C[r,c] = v)
// Double-buffered LDS, one barrier per K-step, branch-free tail staging.
// ---------------------------------------------------------------------------
template<bool TRANSB, bool RELU, bool SCALE_ACC>
__global__ __launch_bounds__(THREADS)
void gemm_wmma(int M, int N, int K,
               const float* __restrict__ A, int lda, long aBatch,
               const float* __restrict__ B, int ldb, long bBatch,
               float* __restrict__ C, int ldc, long cBatch,
               const float* __restrict__ bias,
               const float* __restrict__ addsrc, int ldadd,
               const float* __restrict__ scale, int sstride, int scol,
               float alpha)
{
    __shared__ _Float16 As[2][BM * LDH];
    __shared__ _Float16 Bs[2][BN * LDH];

    const int tid  = threadIdx.x;
    const int z    = blockIdx.z;
    A += (long)z * aBatch;
    B += (long)z * bBatch;
    C += (long)z * cBatch;

    const int mBase = blockIdx.y * BM;
    const int nBase = blockIdx.x * BN;

    const int lane = tid & 31;
    const int wid  = tid >> 5;
    const int wm   = wid & 3;        // 4 row-waves of 32 rows
    const int wn   = wid >> 2;       // 2 col-waves of 64 cols
    const int g    = lane >> 4;      // lane half-group
    const int l16  = lane & 15;

    // staging thread mapping (contiguous 16-half chunks)
    const int rS   = tid >> 1;          // row 0..127
    const int hsS  = (tid & 1) << 4;    // k-half: 0 or 16
    const int kB   = tid & 31;          // normal-B: k within tile
    const int n0B  = (tid >> 5) << 4;   // normal-B: n chunk base

    const bool aAligned = ((((size_t)A) & 15) == 0) && ((lda & 3) == 0);
    const bool bAligned = ((((size_t)B) & 15) == 0) && ((ldb & 3) == 0);
    const bool fullM = (mBase + BM <= M);
    const bool fullN = (nBase + BN <= N);

    v8f acc[2][4];
    #pragma unroll
    for (int mt = 0; mt < 2; ++mt)
        #pragma unroll
        for (int nt = 0; nt < 4; ++nt)
            #pragma unroll
            for (int e = 0; e < 8; ++e) acc[mt][nt][e] = 0.0f;

    // ---- staging helpers (fetch -> regs, store -> LDS) ----
    auto fetchA = [&](Stage16& s, int kt) {
        const int grow = mBase + rS;
        const int gk   = kt + hsS;
        if (fullM && aAligned && (kt + BK <= K)) {
            const float4* p = (const float4*)(A + (long)grow * lda + gk);
            #pragma unroll
            for (int j = 0; j < 4; ++j) {
                float4 f = p[j];
                s.h[4*j+0] = (_Float16)f.x; s.h[4*j+1] = (_Float16)f.y;
                s.h[4*j+2] = (_Float16)f.z; s.h[4*j+3] = (_Float16)f.w;
            }
        } else {                      // branch-free clamped tail path
            const int cr = grow < M ? grow : M - 1;
            #pragma unroll
            for (int j = 0; j < 16; ++j) {
                const int kk = gk + j;
                const int ck = kk < K ? kk : K - 1;
                float v = A[(long)cr * lda + ck];
                s.h[j] = (_Float16)((grow < M && kk < K) ? v : 0.0f);
            }
        }
    };
    auto storeA = [&](const Stage16& s, int b) {
        *(uint4*)&As[b][rS * LDH + hsS]     = s.q[0];
        *(uint4*)&As[b][rS * LDH + hsS + 8] = s.q[1];
    };
    auto fetchB = [&](Stage16& s, int kt) {
        if (TRANSB) {                 // B stored [N,K]: straight copy rows
            const int gn = nBase + rS;
            const int gk = kt + hsS;
            if (fullN && bAligned && (kt + BK <= K)) {
                const float4* p = (const float4*)(B + (long)gn * ldb + gk);
                #pragma unroll
                for (int j = 0; j < 4; ++j) {
                    float4 f = p[j];
                    s.h[4*j+0] = (_Float16)f.x; s.h[4*j+1] = (_Float16)f.y;
                    s.h[4*j+2] = (_Float16)f.z; s.h[4*j+3] = (_Float16)f.w;
                }
            } else {
                const int cn = gn < N ? gn : N - 1;
                #pragma unroll
                for (int j = 0; j < 16; ++j) {
                    const int kk = gk + j;
                    const int ck = kk < K ? kk : K - 1;
                    float v = B[(long)cn * ldb + ck];
                    s.h[j] = (_Float16)((gn < N && kk < K) ? v : 0.0f);
                }
            }
        } else {                      // B stored [K,N]: row of 16 n-values
            const int gk = kt + kB;
            if (fullN && bAligned && gk < K) {
                const float4* p = (const float4*)(B + (long)gk * ldb + nBase + n0B);
                #pragma unroll
                for (int j = 0; j < 4; ++j) {
                    float4 f = p[j];
                    s.h[4*j+0] = (_Float16)f.x; s.h[4*j+1] = (_Float16)f.y;
                    s.h[4*j+2] = (_Float16)f.z; s.h[4*j+3] = (_Float16)f.w;
                }
            } else {
                const int ck = gk < K ? gk : K - 1;
                #pragma unroll
                for (int j = 0; j < 16; ++j) {
                    const int gn = nBase + n0B + j;
                    const int cn = gn < N ? gn : N - 1;
                    float v = B[(long)ck * ldb + cn];
                    s.h[j] = (_Float16)((gk < K && gn < N) ? v : 0.0f);
                }
            }
        }
    };
    auto storeB = [&](const Stage16& s, int b) {
        if (TRANSB) {
            *(uint4*)&Bs[b][rS * LDH + hsS]     = s.q[0];
            *(uint4*)&Bs[b][rS * LDH + hsS + 8] = s.q[1];
        } else {                      // transpose into Bs[n][k]
            #pragma unroll
            for (int j = 0; j < 16; ++j)
                Bs[b][(n0B + j) * LDH + kB] = s.h[j];
        }
    };
    auto compute = [&](int b) {
        v16h afrag[2], bfrag[4];
        #pragma unroll
        for (int mt = 0; mt < 2; ++mt) {
            const int row = wm * 32 + mt * 16 + l16;
            union { v16h v; uint4 q[2]; } u;
            u.q[0] = *(const uint4*)&As[b][row * LDH + 16 * g];       // K: 16g..16g+7
            u.q[1] = *(const uint4*)&As[b][row * LDH + 16 * g + 8];   // K: 16g+8..16g+15
            afrag[mt] = u.v;
        }
        #pragma unroll
        for (int nt = 0; nt < 4; ++nt) {
            const int col = wn * 64 + nt * 16 + l16;
            union { v16h v; uint4 q[2]; } u;
            const uint4* p = (const uint4*)&Bs[b][col * LDH + 16 * g]; // K: 16g..16g+15
            u.q[0] = p[0]; u.q[1] = p[1];
            bfrag[nt] = u.v;
        }
        #pragma unroll
        for (int mt = 0; mt < 2; ++mt)
            #pragma unroll
            for (int nt = 0; nt < 4; ++nt)
                acc[mt][nt] = __builtin_amdgcn_wmma_f32_16x16x32_f16(
                    false, afrag[mt], false, bfrag[nt],
                    (short)0, acc[mt][nt], false, false);
    };

    // ---- prologue: stage tile 0 ----
    {
        Stage16 sa, sb;
        fetchA(sa, 0);
        fetchB(sb, 0);
        storeA(sa, 0);
        storeB(sb, 0);
    }
    // ---- pipelined main loop: fetch(k+1) | compute(k) | store(k+1) ----
    int kt = 0, buf = 0;
    for (;;) {
        __syncthreads();
        const int ktn = kt + BK;
        Stage16 sa, sb;
        const bool more = (ktn < K);
        if (more) { fetchA(sa, ktn); fetchB(sb, ktn); }
        compute(buf);
        if (!more) break;
        storeA(sa, buf ^ 1);
        storeB(sb, buf ^ 1);
        buf ^= 1;
        kt = ktn;
    }

    // ---- epilogue ----
    #pragma unroll
    for (int mt = 0; mt < 2; ++mt) {
        #pragma unroll
        for (int nt = 0; nt < 4; ++nt) {
            const int col = nBase + wn * 64 + nt * 16 + l16;
            if (col >= N) continue;
            #pragma unroll
            for (int i = 0; i < 8; ++i) {
                const int row = mBase + wm * 32 + mt * 16 + 8 * g + i;
                if (row >= M) continue;
                float v = acc[mt][nt][i] * alpha;
                if (bias)   v += bias[col];
                if (addsrc) v += addsrc[(long)row * ldadd + col];
                if (RELU)   v = fmaxf(v, 0.0f);
                const long idx = (long)row * ldc + col;
                if (SCALE_ACC) C[idx] += scale[(long)row * sstride + scol] * v;
                else           C[idx] = v;
            }
        }
    }
}

// ---------------------------------------------------------------------------
// Row L2-norm with eps clamp (matches jnp maximum(norm, 1e-8))
// ---------------------------------------------------------------------------
__global__ __launch_bounds__(256)
void rownorm_kernel(const float* __restrict__ X, int cols, float* __restrict__ out)
{
    __shared__ float red[256];
    const long row = blockIdx.x;
    const float* x = X + row * cols;
    float s = 0.0f;
    for (int c = threadIdx.x; c < cols; c += 256) { float v = x[c]; s += v * v; }
    red[threadIdx.x] = s; __syncthreads();
    for (int off = 128; off > 0; off >>= 1) {
        if (threadIdx.x < off) red[threadIdx.x] += red[threadIdx.x + off];
        __syncthreads();
    }
    if (threadIdx.x == 0) out[row] = fmaxf(sqrtf(red[0]), 1e-8f);
}

// ---------------------------------------------------------------------------
// In-place row softmax (rows contiguous, length cols)
// ---------------------------------------------------------------------------
__global__ __launch_bounds__(256)
void softmax_rows_kernel(float* __restrict__ X, int cols)
{
    __shared__ float red[256];
    const long row = blockIdx.x;
    float* x = X + row * cols;
    float m = -3.0e38f;
    for (int c = threadIdx.x; c < cols; c += 256) m = fmaxf(m, x[c]);
    red[threadIdx.x] = m; __syncthreads();
    for (int off = 128; off > 0; off >>= 1) {
        if (threadIdx.x < off) red[threadIdx.x] = fmaxf(red[threadIdx.x], red[threadIdx.x + off]);
        __syncthreads();
    }
    m = red[0]; __syncthreads();
    float s = 0.0f;
    for (int c = threadIdx.x; c < cols; c += 256) { float e = __expf(x[c] - m); x[c] = e; s += e; }
    red[threadIdx.x] = s; __syncthreads();
    for (int off = 128; off > 0; off >>= 1) {
        if (threadIdx.x < off) red[threadIdx.x] += red[threadIdx.x + off];
        __syncthreads();
    }
    const float inv = 1.0f / red[0];
    for (int c = threadIdx.x; c < cols; c += 256) x[c] *= inv;
}

// ---------------------------------------------------------------------------
// In-place LayerNorm (biased var, eps 1e-5) + ReLU
// ---------------------------------------------------------------------------
__global__ __launch_bounds__(256)
void layernorm_relu_kernel(float* __restrict__ X, const float* __restrict__ gg,
                           const float* __restrict__ bb, int cols)
{
    __shared__ float rs[256], rq[256];
    const long row = blockIdx.x;
    float* x = X + row * cols;
    float s = 0.0f, q = 0.0f;
    for (int c = threadIdx.x; c < cols; c += 256) { float v = x[c]; s += v; q += v * v; }
    rs[threadIdx.x] = s; rq[threadIdx.x] = q; __syncthreads();
    for (int off = 128; off > 0; off >>= 1) {
        if (threadIdx.x < off) { rs[threadIdx.x] += rs[threadIdx.x + off];
                                 rq[threadIdx.x] += rq[threadIdx.x + off]; }
        __syncthreads();
    }
    const float mu   = rs[0] / cols;
    const float var  = rq[0] / cols - mu * mu;
    const float rstd = rsqrtf(var + 1e-5f);
    for (int c = threadIdx.x; c < cols; c += 256) {
        float v = (x[c] - mu) * rstd * gg[c] + bb[c];
        x[c] = fmaxf(v, 0.0f);
    }
}

// ---------------------------------------------------------------------------
// Working memory: cosine*importance -> top-5 -> softmax -> blend of values
// ---------------------------------------------------------------------------
__global__ __launch_bounds__(256)
void wm_finalize_kernel(const float* __restrict__ dots, const float* __restrict__ qn,
                        const float* __restrict__ kn, const float* __restrict__ imp,
                        const float* __restrict__ values,
                        float* __restrict__ out, int M, int D, int ldo)
{
    __shared__ float sv[32 * 5];
    __shared__ int   si[32 * 5];
    __shared__ float w5[5];
    __shared__ int   i5[5];
    const int b = blockIdx.x, tid = threadIdx.x;

    if (tid < 32) {
        float bvv[5]; int bii[5];
        #pragma unroll
        for (int j = 0; j < 5; ++j) { bvv[j] = -3.0e38f; bii[j] = 0; }
        const float invq = 1.0f / qn[b];
        for (int m = tid; m < M; m += 32) {
            float sc = dots[(long)b * M + m] * invq / kn[m] * imp[m];
            if (sc > bvv[4]) {
                int j = 4;
                while (j > 0 && sc > bvv[j - 1]) { bvv[j] = bvv[j - 1]; bii[j] = bii[j - 1]; --j; }
                bvv[j] = sc; bii[j] = m;
            }
        }
        #pragma unroll
        for (int j = 0; j < 5; ++j) { sv[tid * 5 + j] = bvv[j]; si[tid * 5 + j] = bii[j]; }
    }
    __syncthreads();
    if (tid == 0) {
        float bvv[5]; int bii[5];
        #pragma unroll
        for (int j = 0; j < 5; ++j) { bvv[j] = -3.0e38f; bii[j] = 0; }
        for (int c = 0; c < 160; ++c) {
            float sc = sv[c]; int ix = si[c];
            if (sc > bvv[4]) {
                int j = 4;
                while (j > 0 && sc > bvv[j - 1]) { bvv[j] = bvv[j - 1]; bii[j] = bii[j - 1]; --j; }
                bvv[j] = sc; bii[j] = ix;
            }
        }
        const float mx = bvv[0];
        float sum = 0.0f; float e[5];
        #pragma unroll
        for (int j = 0; j < 5; ++j) { e[j] = __expf(bvv[j] - mx); sum += e[j]; }
        #pragma unroll
        for (int j = 0; j < 5; ++j) { w5[j] = e[j] / sum; i5[j] = bii[j]; }
    }
    __syncthreads();
    for (int d = tid; d < D; d += 256) {
        float a = 0.0f;
        #pragma unroll
        for (int j = 0; j < 5; ++j) a += w5[j] * values[(long)i5[j] * D + d];
        out[(long)b * ldo + d] = a;
    }
}

// ---------------------------------------------------------------------------
// Episodic: argmax of cosine, gather that episode representation
// ---------------------------------------------------------------------------
__global__ __launch_bounds__(256)
void episodic_kernel(const float* __restrict__ dots, const float* __restrict__ xn,
                     const float* __restrict__ en, const float* __restrict__ reprs,
                     float* __restrict__ out, int E, int D, int ldo)
{
    __shared__ int sidx;
    const int b = blockIdx.x;
    if (threadIdx.x == 0) {
        float best = -3.0e38f; int bi = 0;
        const float invx = 1.0f / xn[b];
        for (int e = 0; e < E; ++e) {
            float sc = dots[(long)b * E + e] * invx / en[e];
            if (sc > best) { best = sc; bi = e; }
        }
        sidx = bi;
    }
    __syncthreads();
    const int idx = sidx;
    for (int d = threadIdx.x; d < D; d += 256)
        out[(long)b * ldo + d] = reprs[(long)idx * D + d];
}

__global__ __launch_bounds__(256)
void zero_proc_kernel(float* __restrict__ all)
{
    const int i = blockIdx.x * 256 + threadIdx.x;   // 0 .. B*D-1
    const int b = i >> 10, d = i & 1023;
    all[(long)b * 4096 + 3072 + d] = 0.0f;
}

// ---------------------------------------------------------------------------
template<bool TB, bool RELU, bool SACC>
static void launch_gemm(hipStream_t st, int M, int N, int K,
                        const float* A, int lda, long aB,
                        const float* B, int ldb, long bB,
                        float* C, int ldc, long cB,
                        const float* bias, const float* add, int ldadd,
                        const float* scale, int sst, int sc,
                        float alpha, int Z)
{
    dim3 grid((N + BN - 1) / BN, (M + BM - 1) / BM, Z);
    gemm_wmma<TB, RELU, SACC><<<grid, dim3(THREADS), 0, st>>>(
        M, N, K, A, lda, aB, B, ldb, bB, C, ldc, cB, bias, add, ldadd, scale, sst, sc, alpha);
}

extern "C" void kernel_launch(void* const* d_in, const int* in_sizes, int n_in,
                              void* d_out, int out_size, void* d_ws, size_t ws_size,
                              hipStream_t stream)
{
    const int B = 1024, D = 1024, M = 2048, E = 100, C = 1000, S = 50;
    const float ATT_SCALE = 0.0883883476483184f; // 1/sqrt(128)

    const float* x      = (const float*)d_in[0];
    const float* wkeys  = (const float*)d_in[1];
    const float* wvals  = (const float*)d_in[2];
    const float* wimp   = (const float*)d_in[3];
    const float* reprs  = (const float*)d_in[4];
    const float* Wq_wm  = (const float*)d_in[5];
    const float* bq_wm  = (const float*)d_in[6];
    const float* conc   = (const float*)d_in[7];
    const float* Wq     = (const float*)d_in[8];
    const float* bq     = (const float*)d_in[9];
    const float* Wk     = (const float*)d_in[10];
    const float* bk     = (const float*)d_in[11];
    const float* Wv     = (const float*)d_in[12];
    const float* bv     = (const float*)d_in[13];
    const float* Wo     = (const float*)d_in[14];
    const float* bo     = (const float*)d_in[15];
    const float* Wk1    = (const float*)d_in[16];
    const float* bk1    = (const float*)d_in[17];
    const float* ln1_g  = (const float*)d_in[18];
    const float* ln1_b  = (const float*)d_in[19];
    const float* Wk2    = (const float*)d_in[20];
    const float* bk2    = (const float*)d_in[21];
    const float* Wsel   = (const float*)d_in[22];
    const float* bsel   = (const float*)d_in[23];
    const float* Wsk1   = (const float*)d_in[24];
    const float* bsk1   = (const float*)d_in[25];
    const float* Wsk2   = (const float*)d_in[26];
    const float* bsk2   = (const float*)d_in[27];
    const float* Wf1    = (const float*)d_in[28];
    const float* bf1    = (const float*)d_in[29];
    const float* lnf_g  = (const float*)d_in[30];
    const float* lnf_b  = (const float*)d_in[31];
    const float* Wf2    = (const float*)d_in[32];
    const float* bf2    = (const float*)d_in[33];
    (void)in_sizes; (void)n_in; (void)out_size; (void)ws_size;

    float* ws = (float*)d_ws;
    long o = 0;
    float* q_wm   = ws + o; o += (long)B * D;        // 1M
    float* dots   = ws + o; o += (long)B * M;        // 2M
    float* qnorm  = ws + o; o += B;
    float* knorm  = ws + o; o += M;
    float* xnorm  = ws + o; o += B;
    float* enorm  = ws + o; o += 128;
    float* qh     = ws + o; o += (long)B * D;        // 1M
    float* kh     = ws + o; o += (long)C * D + 24576;
    float* vh     = ws + o; o += (long)C * D + 24576;
    float* att    = ws + o; o += (long)B * 8 * C;    // 8.192M
    float* attnd  = ws + o; o += (long)B * D;
    float* comb   = ws + o; o += (long)B * D;
    float* h1     = ws + o; o += (long)B * 2 * D;
    float* allm   = ws + o; o += (long)B * 4 * D;    // 4M (concat buffer)
    float* selw   = ws + o; o += (long)B * S;
    float* hid    = ws + o; o += (long)B * D;
    float* epd    = ws + o; o += (long)B * E;
    float* f1     = ws + o; o += (long)B * 2 * D;

    // ---- Working memory -------------------------------------------------
    launch_gemm<false,false,false>(stream, B, D, D, x, D, 0, Wq_wm, D, 0,
        q_wm, D, 0, bq_wm, nullptr, 0, nullptr, 0, 0, 1.0f, 1);
    launch_gemm<true ,false,false>(stream, B, M, D, q_wm, D, 0, wkeys, D, 0,
        dots, M, 0, nullptr, nullptr, 0, nullptr, 0, 0, 1.0f, 1);
    rownorm_kernel<<<B, 256, 0, stream>>>(q_wm, D, qnorm);
    rownorm_kernel<<<M, 256, 0, stream>>>(wkeys, D, knorm);
    rownorm_kernel<<<B, 256, 0, stream>>>(x, D, xnorm);
    rownorm_kernel<<<E, 256, 0, stream>>>(reprs, D, enorm);
    wm_finalize_kernel<<<B, 256, 0, stream>>>(dots, qnorm, knorm, wimp, wvals,
                                              allm + 0, M, D, 4 * D);

    // ---- Semantic memory: MHA over concepts -----------------------------
    launch_gemm<false,false,false>(stream, B, D, D, x,    D, 0, Wq, D, 0, qh, D, 0, bq, nullptr, 0, nullptr, 0, 0, 1.0f, 1);
    launch_gemm<false,false,false>(stream, C, D, D, conc, D, 0, Wk, D, 0, kh, D, 0, bk, nullptr, 0, nullptr, 0, 0, 1.0f, 1);
    launch_gemm<false,false,false>(stream, C, D, D, conc, D, 0, Wv, D, 0, vh, D, 0, bv, nullptr, 0, nullptr, 0, 0, 1.0f, 1);
    // logits[b,h,c] = (qh_h @ kh_h^T) / sqrt(hd), batched over 8 heads
    launch_gemm<true ,false,false>(stream, B, C, 128, qh, D, 128, kh, D, 128,
        att, 8 * C, C, nullptr, nullptr, 0, nullptr, 0, 0, ATT_SCALE, 8);
    softmax_rows_kernel<<<B * 8, 256, 0, stream>>>(att, C);
    // attended_h = att_h @ vh_h
    launch_gemm<false,false,false>(stream, B, 128, C, att, 8 * C, C, vh, D, 128,
        attnd, D, 128, nullptr, nullptr, 0, nullptr, 0, 0, 1.0f, 8);
    // combined = x + attended @ Wo + bo
    launch_gemm<false,false,false>(stream, B, D, D, attnd, D, 0, Wo, D, 0,
        comb, D, 0, bo, x, D, nullptr, 0, 0, 1.0f, 1);
    launch_gemm<false,false,false>(stream, B, 2 * D, D, comb, D, 0, Wk1, 2 * D, 0,
        h1, 2 * D, 0, bk1, nullptr, 0, nullptr, 0, 0, 1.0f, 1);
    layernorm_relu_kernel<<<B, 256, 0, stream>>>(h1, ln1_g, ln1_b, 2 * D);
    launch_gemm<false,false,false>(stream, B, D, 2 * D, h1, 2 * D, 0, Wk2, D, 0,
        allm + 2 * D, 4 * D, 0, bk2, nullptr, 0, nullptr, 0, 0, 1.0f, 1);

    // ---- Procedural memory: 50 skill MLPs, softmax-weighted --------------
    launch_gemm<false,false,false>(stream, B, S, D, x, D, 0, Wsel, S, 0,
        selw, S, 0, bsel, nullptr, 0, nullptr, 0, 0, 1.0f, 1);
    softmax_rows_kernel<<<B, 256, 0, stream>>>(selw, S);
    zero_proc_kernel<<<(B * D) / 256, 256, 0, stream>>>(allm);
    for (int s = 0; s < S; ++s) {
        launch_gemm<false,true ,false>(stream, B, D, D, x, D, 0,
            Wsk1 + (long)s * D * D, D, 0, hid, D, 0, bsk1 + (long)s * D,
            nullptr, 0, nullptr, 0, 0, 1.0f, 1);
        launch_gemm<false,false,true >(stream, B, D, D, hid, D, 0,
            Wsk2 + (long)s * D * D, D, 0, allm + 3 * D, 4 * D, 0, bsk2 + (long)s * D,
            nullptr, 0, selw, S, s, 1.0f, 1);
    }

    // ---- Episodic memory -------------------------------------------------
    launch_gemm<true ,false,false>(stream, B, E, D, x, D, 0, reprs, D, 0,
        epd, E, 0, nullptr, nullptr, 0, nullptr, 0, 0, 1.0f, 1);
    episodic_kernel<<<B, 256, 0, stream>>>(epd, xnorm, enorm, reprs, allm + D, E, D, 4 * D);

    // ---- Fusion ----------------------------------------------------------
    launch_gemm<false,false,false>(stream, B, 2 * D, 4 * D, allm, 4 * D, 0, Wf1, 2 * D, 0,
        f1, 2 * D, 0, bf1, nullptr, 0, nullptr, 0, 0, 1.0f, 1);
    layernorm_relu_kernel<<<B, 256, 0, stream>>>(f1, lnf_g, lnf_b, 2 * D);
    launch_gemm<false,false,false>(stream, B, D, 2 * D, f1, 2 * D, 0, Wf2, D, 0,
        (float*)d_out, D, 0, bf2, nullptr, 0, nullptr, 0, 0, 1.0f, 1);
}
